// PatchSimCNN_57372173140293
// MI455X (gfx1250) — matrix operand order repaired and checked by reading, not compiled, not run on hardware
//
#include <hip/hip_runtime.h>
#include <stdint.h>

// ---------------------------------------------------------------------------
// PatchSimCNN on MI455X (gfx1250, wave32, WMMA).
//   stage 0: inverse permutation (512 ints)
//   stage 1: pad+convert input to bf16 with 27-px zero halo (no halo checks later)
//   stage 2: pre-swizzle weights into A-fragment-ready bf16 chunks
//   stage 3: implicit-GEMM conv, one wave = 64oc x 32px strip, 8 accumulators,
//            A fragments shared across 2 pixel sub-tiles (halves weight traffic),
//            fused bias+ReLU+permuted scatter (bf16)
//   stage 4: 16x batched 64x64 Gram, one wave = 16x64 strip, 4 accumulators
// ---------------------------------------------------------------------------

typedef __attribute__((ext_vector_type(16))) __bf16         v16bf;
typedef __attribute__((ext_vector_type(16))) unsigned short v16us;
typedef __attribute__((ext_vector_type(8)))  float          v8f;
typedef __attribute__((ext_vector_type(8)))  unsigned int   v8u;

__constant__ int c_fs[8]    = {3, 5, 7, 11, 15, 23, 37, 55};
__constant__ int c_cdx[8]   = {1, 1, 1, 1, 1, 1, 2, 2};          // ceil(k/32)
__constant__ int c_nch[8]   = {9, 15, 21, 33, 45, 69, 222, 330}; // 3*k*ceil(k/32)
__constant__ int c_wbase[8] = {0, 18432, 49152, 92160, 159744, 251904, 393216, 847872};

#define XPAD_W   192          // padded row stride (27-halo + slack)
#define XPAD_H   192
#define XPAD_OFS 27           // halo size = (55-1)/2

// ISA 7.12.2 wave32 fragment K-index maps (16-bit data, 16x16x32).
__device__ __forceinline__ int aK(int v, int half) { return 2 * v + 8 * (half + (v >> 2)); }
// B and packed A here (not ISA order): in-lane K = 2v + e + 16*half  (contiguous run of 16)

__device__ __forceinline__ unsigned short f2bf(float f) {   // RNE f32 -> bf16 bits
    unsigned int u = __builtin_bit_cast(unsigned int, f);
    return (unsigned short)((u + 0x7fffu + ((u >> 16) & 1u)) >> 16);
}

__device__ __forceinline__ v16bf wmma_bf16(v16bf a, v16bf b, v8f& acc);

struct ConvArgs {
    const float* w[8];
    const float* bia[8];
};

// --------------------------- stage 0: inverse perm -------------------------
__global__ void invperm_kernel(const int* __restrict__ perm, int* __restrict__ inv) {
    int c = blockIdx.x * blockDim.x + threadIdx.x;
    if (c < 512) inv[perm[c]] = c;
}

// ------------------ stage 1: pad + convert input to bf16 -------------------
__global__ __launch_bounds__(256)
void pad_input_kernel(const float* __restrict__ x, unsigned short* __restrict__ xpad) {
    const int total = 2 * 3 * XPAD_H * XPAD_W;
    for (int idx = blockIdx.x * blockDim.x + threadIdx.x; idx < total;
         idx += gridDim.x * blockDim.x) {
        const int xx = idx % XPAD_W;
        const int yy = (idx / XPAD_W) % XPAD_H;
        const int bc = idx / (XPAD_W * XPAD_H);      // 0..5 = b*3+c
        const int sx = xx - XPAD_OFS, sy = yy - XPAD_OFS;
        unsigned short v = 0;
        if ((unsigned)sx < 128u && (unsigned)sy < 128u)
            v = f2bf(x[((size_t)bc << 14) + (sy << 7) + sx]);
        xpad[idx] = v;
    }
}

// ----------- stage 2: pre-swizzle weights to A-fragment layout -------------
// Packed element p (per filter): [3:0]=2v+e, [4]=half, rest=(oc*nch + chunk).
// chunk = (c*k + dy)*cdx + dxc ; logical K within chunk = 16*half+2v+e = dx off.
__global__ __launch_bounds__(256)
void prepack_w_kernel(ConvArgs args, unsigned short* __restrict__ wpack) {
    const int f    = blockIdx.y;
    const int k    = c_fs[f];
    const int cdx  = c_cdx[f];
    const int nch  = c_nch[f];
    const int total = 64 * nch * 32;
    const float* __restrict__ wf = args.w[f];
    unsigned short* __restrict__ dst = wpack + c_wbase[f];

    for (int p = blockIdx.x * blockDim.x + threadIdx.x; p < total;
         p += gridDim.x * blockDim.x) {
        const int ve   = p & 15;
        const int half = (p >> 4) & 1;
        const int v = ve >> 1, e = ve & 1;
        const int occ = p >> 5;
        const int cc  = occ % nch;
        const int oc  = occ / nch;
        const int dxc = cc % cdx;
        const int row = cc / cdx;
        const int dy  = row % k;
        const int c   = row / k;
        const int dx  = (dxc << 5) + 16 * half + 2 * v + e;
        unsigned short val = 0;
        if (dx < k)
            val = f2bf(wf[(((size_t)oc * 3 + c) * k + dy) * k + dx]);
        dst[p] = val;
    }
}

// ------------------- stage 3: implicit-GEMM conv (WMMA) --------------------
// One wave = 64(oc) x 32(pixel) strip of one filter branch.
// Per K-chunk: 4 fragment-ready A loads + 2 B (im2col) fragments -> 8 WMMAs.
// wave id bits: [8:0]=pixel tile (32px), [9]=batch, [12:10]=filter.
__global__ __launch_bounds__(256)
void conv_wmma_kernel(const unsigned short* __restrict__ xpad,
                      const unsigned short* __restrict__ wpack,
                      ConvArgs args,
                      const int* __restrict__ invperm,
                      unsigned short* __restrict__ feat) {
    const int wave = (blockIdx.x << 3) | (threadIdx.x >> 5);   // 0..8191
    const int lane = threadIdx.x & 31;
    const int half = lane >> 4;
    const int l16  = lane & 15;

    const int ptile = wave & 511;
    const int b     = (wave >> 9) & 1;
    const int f     = wave >> 10;

    const int k   = c_fs[f];
    const int cdx = c_cdx[f];
    const int nch = c_nch[f];
    const int pad = (k - 1) >> 1;

    // Two pixel sub-tiles: columns p0 (this lane) and p0+16.
    const int p0  = (ptile << 5) | l16;
    const int py0 = p0 >> 7,        px0 = p0 & 127;
    const int p1  = p0 + 16;
    const int py1 = p1 >> 7,        px1 = p1 & 127;

    // packed A dword bases: per oc-tile, lane row = t*16+l16, this half
    const unsigned int* __restrict__ wpu = (const unsigned int*)(wpack + c_wbase[f]);
    const unsigned int* ap0 = wpu + ((size_t)(0 * 16 + l16) * nch) * 16 + (half << 3);
    const unsigned int* ap1 = wpu + ((size_t)(1 * 16 + l16) * nch) * 16 + (half << 3);
    const unsigned int* ap2 = wpu + ((size_t)(2 * 16 + l16) * nch) * 16 + (half << 3);
    const unsigned int* ap3 = wpu + ((size_t)(3 * 16 + l16) * nch) * 16 + (half << 3);

    // padded input: per-lane row starts (halo makes all accesses in-bounds)
    const unsigned short* __restrict__ xbase =
        xpad + ((size_t)(b * 3) * XPAD_H) * XPAD_W + (half << 4);
    const unsigned short* xb0 = xbase + (size_t)(py0 - pad + XPAD_OFS) * XPAD_W
                                      + (px0 - pad + XPAD_OFS);
    const unsigned short* xb1 = xbase + (size_t)(py1 - pad + XPAD_OFS) * XPAD_W
                                      + (px1 - pad + XPAD_OFS);

    v8f acc00 = {}, acc10 = {}, acc20 = {}, acc30 = {};  // pixel sub-tile 0
    v8f acc01 = {}, acc11 = {}, acc21 = {}, acc31 = {};  // pixel sub-tile 1
    int cc = 0;
    for (int c = 0; c < 3; ++c) {
        for (int dy = 0; dy < k; ++dy) {
            const size_t rofs = ((size_t)c * XPAD_H + dy) * XPAD_W;
            const unsigned short* __restrict__ bs0 = xb0 + rofs;
            const unsigned short* __restrict__ bs1 = xb1 + rofs;
            for (int dxc = 0; dxc < cdx; ++dxc, ++cc) {
                const size_t co = (size_t)cc << 4;
                __builtin_prefetch(ap0 + co + 16, 0, 0);    // global_prefetch_b8
                // ---- B: two 16-wide contiguous bf16 runs (one per sub-tile)
                v16us bw0, bw1;
                __builtin_memcpy(&bw0, bs0 + (dxc << 5), 32);
                __builtin_memcpy(&bw1, bs1 + (dxc << 5), 32);
                const v16bf b0 = __builtin_bit_cast(v16bf, bw0);
                const v16bf b1 = __builtin_bit_cast(v16bf, bw1);
                // ---- A: 4 fragment-ready 32B vector loads, shared by b0/b1
                const v16bf a0 = __builtin_bit_cast(v16bf, *(const v8u*)(ap0 + co));
                const v16bf a1 = __builtin_bit_cast(v16bf, *(const v8u*)(ap1 + co));
                const v16bf a2 = __builtin_bit_cast(v16bf, *(const v8u*)(ap2 + co));
                const v16bf a3 = __builtin_bit_cast(v16bf, *(const v8u*)(ap3 + co));
                // ---- 8 independent WMMAs (no D->A/B hazards between them)
                acc00 = __builtin_amdgcn_wmma_f32_16x16x32_bf16(false, a0, false, b0,
                                                                (short)0, acc00, false, false);
                acc10 = __builtin_amdgcn_wmma_f32_16x16x32_bf16(false, a1, false, b0,
                                                                (short)0, acc10, false, false);
                acc20 = __builtin_amdgcn_wmma_f32_16x16x32_bf16(false, a2, false, b0,
                                                                (short)0, acc20, false, false);
                acc30 = __builtin_amdgcn_wmma_f32_16x16x32_bf16(false, a3, false, b0,
                                                                (short)0, acc30, false, false);
                acc01 = __builtin_amdgcn_wmma_f32_16x16x32_bf16(false, a0, false, b1,
                                                                (short)0, acc01, false, false);
                acc11 = __builtin_amdgcn_wmma_f32_16x16x32_bf16(false, a1, false, b1,
                                                                (short)0, acc11, false, false);
                acc21 = __builtin_amdgcn_wmma_f32_16x16x32_bf16(false, a2, false, b1,
                                                                (short)0, acc21, false, false);
                acc31 = __builtin_amdgcn_wmma_f32_16x16x32_bf16(false, a3, false, b1,
                                                                (short)0, acc31, false, false);
            }
        }
    }

    // Epilogue: bias + ReLU + permuted channel scatter
    // C layout: VGPR j -> M = j + 8*half, N = lane&15.
    const float* __restrict__ bias = args.bia[f];
    v8f accs[2][4] = {{acc00, acc10, acc20, acc30}, {acc01, acc11, acc21, acc31}};
#pragma unroll
    for (int u = 0; u < 2; ++u) {
        const int pu = (u == 0) ? p0 : p1;
#pragma unroll
        for (int t = 0; t < 4; ++t) {
#pragma unroll
            for (int j = 0; j < 8; ++j) {
                const int oc = (t << 4) + (half << 3) + j;
                float v = accs[u][t][j] + bias[oc];
                v = v > 0.0f ? v : 0.0f;
                const int cnew = invperm[(f << 6) + oc];
                feat[(((size_t)b * 512 + cnew) << 14) + pu] = f2bf(v);
            }
        }
    }
}

// ------------------------------ stage 4: Gram ------------------------------
// 16 groups (b*8+g), each 64x16384 bf16 -> 64x64 f32 Gram.
// One wave = one 16x64 strip (4 tiles): A fragment loaded once per K-chunk,
// 4 B fragments, 4 WMMAs. Fragments = dword loads of packed bf16 pairs.
__global__ __launch_bounds__(256)
void gram_wmma_kernel(const unsigned short* __restrict__ feat,
                      float* __restrict__ out) {
    const int wave = (blockIdx.x << 3) | (threadIdx.x >> 5);   // 0..63
    const int lane = threadIdx.x & 31;
    const int half = lane >> 4;
    const int l16  = lane & 15;

    const int mt  = wave & 3;     // row strip of the 64x64 gram
    const int grp = wave >> 2;    // 0..15 == b*8+g

    const unsigned int* __restrict__ fp = (const unsigned int*)feat;
    const size_t ubase = ((size_t)grp << 19);           // grp*64*16384/2 dwords
    const size_t uA  = ubase + ((size_t)((mt << 4) + l16) << 13);
    const size_t uB0 = ubase + ((size_t)(0 * 16 + l16) << 13);
    const size_t uB1 = ubase + ((size_t)(1 * 16 + l16) << 13);
    const size_t uB2 = ubase + ((size_t)(2 * 16 + l16) << 13);
    const size_t uB3 = ubase + ((size_t)(3 * 16 + l16) << 13);

    v8f acc0 = {}, acc1 = {}, acc2 = {}, acc3 = {};
    for (int kk = 0; kk < 16384; kk += 32) {
        v8u ua, ub0, ub1, ub2, ub3;
#pragma unroll
        for (int v = 0; v < 8; ++v) {
            const size_t ka = (size_t)(kk + 2 * v + 16 * half) >> 1;
            const size_t kb = (size_t)(kk + 2 * v + 16 * half) >> 1;
            ua[v]  = fp[uA + ka];
            ub0[v] = fp[uB0 + kb];
            ub1[v] = fp[uB1 + kb];
            ub2[v] = fp[uB2 + kb];
            ub3[v] = fp[uB3 + kb];
        }
        const v16bf af = __builtin_bit_cast(v16bf, ua);
        acc0 = __builtin_amdgcn_wmma_f32_16x16x32_bf16(
            false, af, false, __builtin_bit_cast(v16bf, ub0), (short)0, acc0, false, false);
        acc1 = __builtin_amdgcn_wmma_f32_16x16x32_bf16(
            false, af, false, __builtin_bit_cast(v16bf, ub1), (short)0, acc1, false, false);
        acc2 = __builtin_amdgcn_wmma_f32_16x16x32_bf16(
            false, af, false, __builtin_bit_cast(v16bf, ub2), (short)0, acc2, false, false);
        acc3 = __builtin_amdgcn_wmma_f32_16x16x32_bf16(
            false, af, false, __builtin_bit_cast(v16bf, ub3), (short)0, acc3, false, false);
    }

    v8f accs[4] = {acc0, acc1, acc2, acc3};
#pragma unroll
    for (int nt = 0; nt < 4; ++nt) {
#pragma unroll
        for (int j = 0; j < 8; ++j) {
            const int m = (mt << 4) + (half << 3) + j;
            const int n = (nt << 4) + l16;
            out[(size_t)grp * 4096 + m * 64 + n] = accs[nt][j];
        }
    }
}

// ------------------------------- launcher ----------------------------------
extern "C" void kernel_launch(void* const* d_in, const int* in_sizes, int n_in,
                              void* d_out, int out_size, void* d_ws, size_t ws_size,
                              hipStream_t stream) {
    const float* x = (const float*)d_in[0];
    ConvArgs args;
    for (int i = 0; i < 8; ++i) {
        args.w[i]   = (const float*)d_in[1 + 2 * i];
        args.bia[i] = (const float*)d_in[2 + 2 * i];
    }
    const int* perm = (const int*)d_in[17];

    char* ws = (char*)d_ws;
    int*            inv   = (int*)ws;                             // 2 KB
    unsigned short* xpad  = (unsigned short*)(ws + 4096);         // 432 KB
    unsigned short* wpack = (unsigned short*)(ws + 458752);       // ~3 MB
    unsigned short* feat  = (unsigned short*)(ws + 4194304);      // 32 MB
    float* out = (float*)d_out;

    invperm_kernel<<<2, 256, 0, stream>>>(perm, inv);
    pad_input_kernel<<<864, 256, 0, stream>>>(x, xpad);
    prepack_w_kernel<<<dim3(2640, 8), 256, 0, stream>>>(args, wpack);
    // 8 filters * 2 batches * 512 pixel-tiles = 8192 waves (64oc x 32px each)
    conv_wmma_kernel<<<1024, 256, 0, stream>>>(xpad, wpack, args, inv, feat);
    // 16 groups * 4 row-strips = 64 waves (16x64 each)
    gram_wmma_kernel<<<8, 256, 0, stream>>>(feat, out);
}
